// GCNBlock_73667279061347
// MI455X (gfx1250) — matrix operand order repaired and checked by reading, not compiled, not run on hardware
//
#include <hip/hip_runtime.h>

// ---------------- CDNA5 WMMA types ----------------
typedef __bf16 bf16;
typedef __attribute__((ext_vector_type(16))) __bf16 v16bf;
typedef __attribute__((ext_vector_type(8)))  float  v8f;

#define D_NODE 256
#define D_EDGE 64
#define HEADS 32
#define QKVS_LD 1024   // q|k|v|skip concatenated per node

// =====================================================================
// Weight pre-pack: convert fp32 weights to bf16 in WMMA B-fragment order.
// Packed layout: [tile_n][k_step][lane(32)][elem(16)] so a wave fetches its
// whole 32x16 B fragment with one 32B-per-lane coalesced load.
// B layout (dense 16-bit 32x16): lane half h, elem s -> K=8h+s for s<8,
// K=16+8h+(s-8) for s>=8 (matches A fragment order).
// =====================================================================
__device__ inline void pack_frag_dev(const float* __restrict__ W, int K, int Nout,
                                     bf16* __restrict__ out, int tid, int nthr) {
  const int ks_count = K >> 5;
  const int total = (Nout >> 4) * ks_count * 512;
  for (int i = tid; i < total; i += nthr) {
    int elem = i & 15;
    int lane = (i >> 4) & 31;
    int blk  = i >> 9;
    int ks = blk % ks_count;
    int tn = blk / ks_count;
    int col = (tn << 4) + (lane & 15);
    int k   = (ks << 5) + ((lane >> 4) << 3) + (elem & 7) + ((elem >> 3) << 4);
    out[i] = (bf16)W[(size_t)k * Nout + col];
  }
}

__global__ __launch_bounds__(256) void pack_weights(
    const float* Wq, const float* Wk, const float* Wv, const float* Ws,
    const float* bq, const float* bk, const float* bv, const float* bs,
    const float* We, const float* W1, const float* W2,
    bf16* Wnode, float* bnode, bf16* Wep, bf16* W1p, bf16* W2p) {
  int tid = blockIdx.x * blockDim.x + threadIdx.x;
  int nthr = gridDim.x * blockDim.x;
  // Wq|Wk|Wv|Ws -> one 256x1024 packed matrix (each 256x256 = 65536 packed elems)
  pack_frag_dev(Wq, 256, 256, Wnode + 0 * 65536, tid, nthr);
  pack_frag_dev(Wk, 256, 256, Wnode + 1 * 65536, tid, nthr);
  pack_frag_dev(Wv, 256, 256, Wnode + 2 * 65536, tid, nthr);
  pack_frag_dev(Ws, 256, 256, Wnode + 3 * 65536, tid, nthr);
  pack_frag_dev(We, 64, 256, Wep, tid, nthr);
  pack_frag_dev(W1, 576, 64, W1p, tid, nthr);
  pack_frag_dev(W2, 64, 64, W2p, tid, nthr);
  for (int i = tid; i < 1024; i += nthr) {
    const float* bsrc = (i < 256) ? bq : (i < 512) ? bk : (i < 768) ? bv : bs;
    bnode[i] = bsrc[i & 255];
  }
}

__global__ __launch_bounds__(256) void init_kernel(float* msg, float* amax, float* denom,
                                                   long nmsg, long nh) {
  long i = (long)blockIdx.x * blockDim.x + threadIdx.x;
  long stride = (long)gridDim.x * blockDim.x;
  for (long j = i; j < nmsg; j += stride) msg[j] = 0.f;
  for (long j = i; j < nh; j += stride) {
    amax[j]  = __uint_as_float(0xFF800000u);  // -inf
    denom[j] = 0.f;
  }
}

// =====================================================================
// Generic fp32-in / bf16-WMMA / fp32-out GEMM: C[M][ldc] = A[M][K] @ B + bias
// One wave = one 16x16 tile, 8 waves/block = 128 output columns.
// A staged via LDS (stride K+8 to avoid bank conflicts), B from packed frags.
// =====================================================================
__global__ __launch_bounds__(256) void wmma_gemm(
    const float* __restrict__ A, int M, int K,
    const bf16* __restrict__ Bp, const float* __restrict__ bias,
    float* __restrict__ C, int ldc) {
  __shared__ bf16 sA[16 * 264];
  const int ksteps = K >> 5;
  const int strideA = K + 8;
  const int m0 = blockIdx.x << 4;
  const int wave = threadIdx.x >> 5;
  const int lane = threadIdx.x & 31;
  const int n0 = (blockIdx.y << 7) + (wave << 4);

  for (int i = threadIdx.x; i < 16 * K; i += 256) {
    int r = i / K, c = i - r * K;
    int gm = m0 + r;
    float v = (gm < M) ? A[(size_t)gm * K + c] : 0.f;
    sA[r * strideA + c] = (bf16)v;
  }
  __syncthreads();

  const int row = lane & 15;
  const int kbA = (lane >> 4) << 3;  // A: lanes 16-31 offset K by +8
  const bf16* fragB = Bp + (((size_t)(n0 >> 4) * ksteps) << 9) + ((size_t)lane << 4);

  v8f acc = {};
  for (int ks = 0; ks < ksteps; ++ks) {
    v16bf a;
    const int kk = ks << 5;
#pragma unroll
    for (int p = 0; p < 8; ++p) {
      int ka = kk + kbA + ((p < 4) ? (p << 1) : (16 + ((p - 4) << 1)));
      a[2 * p]     = sA[row * strideA + ka];
      a[2 * p + 1] = sA[row * strideA + ka + 1];
    }
    v16bf b = *(const v16bf*)(fragB + ((size_t)ks << 9));
    acc = __builtin_amdgcn_wmma_f32_16x16x32_bf16(false, a, false, b, (short)0, acc,
                                                  false, false);
  }

  const int gn = n0 + (lane & 15);
  const float bn = bias ? bias[gn] : 0.f;
  const int mo = (lane >> 4) << 3;  // D: lanes 16-31 hold rows M+8
#pragma unroll
  for (int r = 0; r < 8; ++r) {
    int gm = m0 + mo + r;
    if (gm < M) C[(size_t)gm * ldc + gn] = acc[r] + bn;
  }
}

// ---------------- attention (scatter-softmax, L2-resident atomics) ----------------
__device__ inline void atomicMaxFloat(float* addr, float val) {
  if (!(val < 0.0f)) atomicMax((int*)addr, __float_as_int(val));
  else               atomicMin((unsigned int*)addr, __float_as_uint(val));
}

__global__ __launch_bounds__(256) void attn_score(
    const float* __restrict__ qkvs, const float* __restrict__ eproj,
    const int* __restrict__ src, const int* __restrict__ dst,
    float* __restrict__ alpha, float* __restrict__ amax, int EH) {
  int t = blockIdx.x * blockDim.x + threadIdx.x;
  if (t >= EH) return;
  int e = t >> 5, h = t & 31;
  int s = src[e], d = dst[e];
  const float4* q4 = (const float4*)(qkvs + (size_t)d * QKVS_LD + h * 8);        // q
  const float4* k4 = (const float4*)(qkvs + (size_t)s * QKVS_LD + 256 + h * 8);  // k
  const float4* e4 = (const float4*)(eproj + (size_t)e * 256 + h * 8);
  float4 q0 = q4[0], q1 = q4[1], k0 = k4[0], k1 = k4[1], e0 = e4[0], e1 = e4[1];
  float acc = q0.x * (k0.x + e0.x) + q0.y * (k0.y + e0.y) + q0.z * (k0.z + e0.z) +
              q0.w * (k0.w + e0.w) + q1.x * (k1.x + e1.x) + q1.y * (k1.y + e1.y) +
              q1.z * (k1.z + e1.z) + q1.w * (k1.w + e1.w);
  acc *= 0.35355339059327373f;  // 1/sqrt(HEAD_DIM)
  alpha[t] = acc;
  atomicMaxFloat(&amax[(size_t)d * HEADS + h], acc);
}

__global__ __launch_bounds__(256) void attn_exp(
    float* __restrict__ alpha, const float* __restrict__ amax,
    float* __restrict__ denom, const int* __restrict__ dst, int EH) {
  int t = blockIdx.x * blockDim.x + threadIdx.x;
  if (t >= EH) return;
  int e = t >> 5, h = t & 31;
  int d = dst[e];
  float ex = __expf(alpha[t] - amax[(size_t)d * HEADS + h]);
  alpha[t] = ex;
  atomicAdd(&denom[(size_t)d * HEADS + h], ex);
}

__global__ __launch_bounds__(256) void attn_msg(
    const float* __restrict__ qkvs, const float* __restrict__ eproj,
    const float* __restrict__ alpha, const float* __restrict__ denom,
    const int* __restrict__ src, const int* __restrict__ dst,
    float* __restrict__ msg, int EH) {
  int t = blockIdx.x * blockDim.x + threadIdx.x;
  if (t >= EH) return;
  int e = t >> 5, h = t & 31;
  int s = src[e], d = dst[e];
  float att = alpha[t] / (denom[(size_t)d * HEADS + h] + 1e-16f);
  const float4* v4 = (const float4*)(qkvs + (size_t)s * QKVS_LD + 512 + h * 8);  // v
  const float4* e4 = (const float4*)(eproj + (size_t)e * 256 + h * 8);
  float4 v0 = v4[0], v1 = v4[1], e0 = e4[0], e1 = e4[1];
  float* mp = msg + (size_t)d * D_NODE + h * 8;
  atomicAdd(mp + 0, (v0.x + e0.x) * att);
  atomicAdd(mp + 1, (v0.y + e0.y) * att);
  atomicAdd(mp + 2, (v0.z + e0.z) * att);
  atomicAdd(mp + 3, (v0.w + e0.w) * att);
  atomicAdd(mp + 4, (v1.x + e1.x) * att);
  atomicAdd(mp + 5, (v1.y + e1.y) * att);
  atomicAdd(mp + 6, (v1.z + e1.z) * att);
  atomicAdd(mp + 7, (v1.w + e1.w) * att);
}

__global__ __launch_bounds__(256) void node_finalize(
    const float* __restrict__ msg, const float* __restrict__ qkvs,
    float* __restrict__ xout, long total) {
  long t = (long)blockIdx.x * blockDim.x + threadIdx.x;
  if (t >= total) return;
  long row = t >> 8;
  int col = (int)(t & 255);
  float v = msg[t] + qkvs[row * QKVS_LD + 768 + col];  // + skip (x@Ws+bs)
  xout[t] = (v >= 0.f) ? v : 0.01f * v;                // block leaky-relu
}

// =====================================================================
// Fused EdgeResidualLayer: gather [x_new[src]|x_new[dst]|edge_attr] (16x576)
// into LDS, GEMM1 (K=576, 18 WMMA steps) -> leaky -> bf16 h in LDS ->
// GEMM2 (K=64, 2 WMMA steps) -> residual + leaky -> store. 4 waves cover
// the 64 output columns.
// =====================================================================
__global__ __launch_bounds__(128) void edge_mlp(
    const float* __restrict__ xn, const float* __restrict__ ea,
    const int* __restrict__ src, const int* __restrict__ dst,
    const bf16* __restrict__ W1p, const float* __restrict__ b1,
    const bf16* __restrict__ W2p, const float* __restrict__ b2,
    float* __restrict__ out2, int E) {
  __shared__ bf16 sIn[16 * 584];
  __shared__ bf16 sH[16 * 72];
  const int m0 = blockIdx.x << 4;
  const int wave = threadIdx.x >> 5;
  const int lane = threadIdx.x & 31;

  for (int i = threadIdx.x; i < 16 * 576; i += 128) {
    int r = i / 576, c = i - r * 576;
    int e = m0 + r;
    float v = 0.f;
    if (e < E) {
      if (c < 256)      v = xn[(size_t)src[e] * D_NODE + c];
      else if (c < 512) v = xn[(size_t)dst[e] * D_NODE + (c - 256)];
      else              v = ea[(size_t)e * D_EDGE + (c - 512)];
    }
    sIn[r * 584 + c] = (bf16)v;
  }
  __syncthreads();

  const int row = lane & 15;
  const int kbA = (lane >> 4) << 3;
  const int n0 = wave << 4;
  const int gn = n0 + (lane & 15);
  const int mo = (lane >> 4) << 3;

  v8f acc = {};
  {
    const bf16* fragB = W1p + (((size_t)wave * 18) << 9) + ((size_t)lane << 4);
    for (int ks = 0; ks < 18; ++ks) {
      v16bf a;
      const int kk = ks << 5;
#pragma unroll
      for (int p = 0; p < 8; ++p) {
        int ka = kk + kbA + ((p < 4) ? (p << 1) : (16 + ((p - 4) << 1)));
        a[2 * p]     = sIn[row * 584 + ka];
        a[2 * p + 1] = sIn[row * 584 + ka + 1];
      }
      v16bf b = *(const v16bf*)(fragB + ((size_t)ks << 9));
      acc = __builtin_amdgcn_wmma_f32_16x16x32_bf16(false, a, false, b, (short)0, acc,
                                                    false, false);
    }
  }
  const float bn1 = b1[gn];
#pragma unroll
  for (int r = 0; r < 8; ++r) {
    float h = acc[r] + bn1;
    h = (h >= 0.f) ? h : 0.01f * h;
    sH[(mo + r) * 72 + gn] = (bf16)h;
  }
  __syncthreads();

  v8f acc2 = {};
  {
    const bf16* fragB = W2p + (((size_t)wave * 2) << 9) + ((size_t)lane << 4);
    for (int ks = 0; ks < 2; ++ks) {
      v16bf a;
      const int kk = ks << 5;
#pragma unroll
      for (int p = 0; p < 8; ++p) {
        int ka = kk + kbA + ((p < 4) ? (p << 1) : (16 + ((p - 4) << 1)));
        a[2 * p]     = sH[row * 72 + ka];
        a[2 * p + 1] = sH[row * 72 + ka + 1];
      }
      v16bf b = *(const v16bf*)(fragB + ((size_t)ks << 9));
      acc2 = __builtin_amdgcn_wmma_f32_16x16x32_bf16(false, a, false, b, (short)0, acc2,
                                                     false, false);
    }
  }
  const float bn2 = b2[gn];
#pragma unroll
  for (int r = 0; r < 8; ++r) {
    int e = m0 + mo + r;
    if (e < E) {
      float v = ea[(size_t)e * D_EDGE + gn] + acc2[r] + bn2;  // residual
      out2[(size_t)e * D_EDGE + gn] = (v >= 0.f) ? v : 0.01f * v;
    }
  }
}

// =====================================================================
extern "C" void kernel_launch(void* const* d_in, const int* in_sizes, int n_in,
                              void* d_out, int out_size, void* d_ws, size_t ws_size,
                              hipStream_t stream) {
  const float* x  = (const float*)d_in[0];
  const int*   ei = (const int*)d_in[1];
  const float* ea = (const float*)d_in[2];
  const float* Wq = (const float*)d_in[3];
  const float* bq = (const float*)d_in[4];
  const float* Wk = (const float*)d_in[5];
  const float* bk = (const float*)d_in[6];
  const float* Wv = (const float*)d_in[7];
  const float* bv = (const float*)d_in[8];
  const float* We = (const float*)d_in[9];
  const float* Ws = (const float*)d_in[10];
  const float* bs = (const float*)d_in[11];
  const float* W1 = (const float*)d_in[12];
  const float* b1 = (const float*)d_in[13];
  const float* W2 = (const float*)d_in[14];
  const float* b2 = (const float*)d_in[15];

  const int N = in_sizes[0] / D_NODE;
  const int E = in_sizes[2] / D_EDGE;
  const int* src = ei;
  const int* dst = ei + E;

  // ---- workspace carve-up ----
  char* p = (char*)d_ws;
  bf16* Wnode = (bf16*)p; p += (size_t)4 * 65536 * sizeof(bf16);   // 256x1024 packed
  float* bnode = (float*)p; p += 1024 * sizeof(float);
  bf16* Wep = (bf16*)p; p += (size_t)16 * 2 * 512 * sizeof(bf16);  // 64x256 packed
  bf16* W1p = (bf16*)p; p += (size_t)4 * 18 * 512 * sizeof(bf16);  // 576x64 packed
  bf16* W2p = (bf16*)p; p += (size_t)4 * 2 * 512 * sizeof(bf16);   // 64x64 packed
  float* qkvs  = (float*)p; p += (size_t)N * QKVS_LD * sizeof(float);
  float* eproj = (float*)p; p += (size_t)E * 256 * sizeof(float);
  float* alpha = (float*)p; p += (size_t)E * HEADS * sizeof(float);
  float* amax  = (float*)p; p += (size_t)N * HEADS * sizeof(float);
  float* denom = (float*)p; p += (size_t)N * HEADS * sizeof(float);
  float* msg   = (float*)p; p += (size_t)N * D_NODE * sizeof(float);
  if ((size_t)(p - (char*)d_ws) > ws_size) return;  // scratch too small

  float* xn_out = (float*)d_out;
  float* en_out = xn_out + (size_t)N * D_NODE;

  pack_weights<<<256, 256, 0, stream>>>(Wq, Wk, Wv, Ws, bq, bk, bv, bs, We, W1, W2,
                                        Wnode, bnode, Wep, W1p, W2p);
  init_kernel<<<2048, 256, 0, stream>>>(msg, amax, denom, (long)N * D_NODE,
                                        (long)N * HEADS);

  dim3 gn_((N + 15) / 16, 8);  // 1024 output cols = q|k|v|skip
  wmma_gemm<<<gn_, 256, 0, stream>>>(x, N, 256, Wnode, bnode, qkvs, QKVS_LD);
  dim3 ge_((E + 15) / 16, 2);  // 256 output cols, no bias (lin_edge)
  wmma_gemm<<<ge_, 256, 0, stream>>>(ea, E, 64, Wep, nullptr, eproj, 256);

  const int EH = E * HEADS;
  const int gEH = (EH + 255) / 256;
  attn_score<<<gEH, 256, 0, stream>>>(qkvs, eproj, src, dst, alpha, amax, EH);
  attn_exp<<<gEH, 256, 0, stream>>>(alpha, amax, denom, dst, EH);
  attn_msg<<<gEH, 256, 0, stream>>>(qkvs, eproj, alpha, denom, src, dst, msg, EH);

  const long tot = (long)N * D_NODE;
  node_finalize<<<(int)((tot + 255) / 256), 256, 0, stream>>>(msg, qkvs, xn_out, tot);

  edge_mlp<<<(E + 15) / 16, 128, 0, stream>>>(xn_out, ea, src, dst, W1p, b1, W2p, b2,
                                              en_out, E);
}
